// TransformerModel_7275674599851
// MI455X (gfx1250) — matrix operand order repaired and checked by reading, not compiled, not run on hardware
//
#include <hip/hip_runtime.h>
#include <hip/hip_bf16.h>
#include <stdint.h>

typedef _Float16 h16;
typedef __attribute__((ext_vector_type(16))) _Float16 v16h;
typedef __attribute__((ext_vector_type(8)))  _Float16 v8h;
typedef __attribute__((ext_vector_type(8)))  float    v8f;
typedef int v4i_g __attribute__((vector_size(16)));   // matches builtin param pointee

#define NN    24576
#define BG    128
#define DIN   92
#define DINP  96
#define DD    512
#define HH    8
#define HDIM  64
#define LL    3
#define DFFN  2048
#define SMAX  256
#define MTOT  (BG*SMAX)

#define APITCH 40   // LDS row pitch in halves (80B = 5x16B, conflict-free)

#if defined(__has_builtin)
#if __has_builtin(__builtin_amdgcn_global_load_async_to_lds_b128)
#define HAS_ASYNC_LDS 1
#endif
#endif

static inline int cdiv(long a, long b) { return (int)((a + b - 1) / b); }

// 16-byte global -> LDS copy (async DMA path when available)
__device__ __forceinline__ void lds_copy16(const h16* g, h16* l) {
#ifdef HAS_ASYNC_LDS
  __builtin_amdgcn_global_load_async_to_lds_b128(
      (__attribute__((address_space(1))) v4i_g*)g,
      (__attribute__((address_space(3))) v4i_g*)(uint32_t)(uintptr_t)l, 0, 0);
#else
  *(v8h*)l = *(const v8h*)g;
#endif
}

__device__ __forceinline__ void lds_copy_join() {
#ifdef HAS_ASYNC_LDS
#if __has_builtin(__builtin_amdgcn_s_wait_asynccnt)
  __builtin_amdgcn_s_wait_asynccnt(0);
#else
  asm volatile("s_wait_asynccnt 0x0" ::: "memory");
#endif
#endif
}

// ---------------- utility kernels ----------------

__global__ void k_fill_u32(uint32_t* p, uint32_t v, long n) {
  long i = (long)blockIdx.x * blockDim.x + threadIdx.x;
  if (i < n) p[i] = v;
}

__global__ void k_cvt_f32_f16(const float* __restrict__ s, h16* __restrict__ d, long n) {
  long i = (long)blockIdx.x * blockDim.x + threadIdx.x;
  if (i < n) d[i] = (h16)s[i];
}

// pad [R, DIN] f32 -> [R, DINP] f16 (zero-padded K)
__global__ void k_pad96(const float* __restrict__ s, h16* __restrict__ d, long rows) {
  long i = (long)blockIdx.x * blockDim.x + threadIdx.x;
  long n = rows * DINP;
  if (i >= n) return;
  int kcol = (int)(i % DINP);
  long r = i / DINP;
  d[i] = (kcol < DIN) ? (h16)s[r * DIN + kcol] : (h16)0.f;
}

__global__ void k_counts(const int* __restrict__ batch, int* __restrict__ counts) {
  int i = blockIdx.x * blockDim.x + threadIdx.x;
  if (i < NN) atomicAdd(&counts[batch[i]], 1);
}

__global__ void k_starts(const int* __restrict__ counts, int* __restrict__ starts) {
  if (threadIdx.x == 0) {
    int acc = 0;
    for (int b = 0; b < BG; ++b) { starts[b] = acc; acc += counts[b]; }
  }
}

__global__ void k_pos(const int* __restrict__ batch, const int* __restrict__ starts,
                      int* __restrict__ pos) {
  int i = blockIdx.x * blockDim.x + threadIdx.x;
  if (i < NN) pos[i] = i - starts[batch[i]];
}

// scatter h[N,512] f32 into padded hpF/hpH (B*SMAX, 512)
__global__ void k_scatter(const float* __restrict__ h, const int* __restrict__ batch,
                          const int* __restrict__ pos, float* __restrict__ hpF,
                          h16* __restrict__ hpH) {
  int n = blockIdx.x, t = threadIdx.x;
  int b = batch[n], p = pos[n];
  if (p >= SMAX) return;
  long src = (long)n * DD, dst = ((long)b * SMAX + p) * DD;
  for (int d = t; d < DD; d += 256) {
    float v = h[src + d];
    hpF[dst + d] = v;
    hpH[dst + d] = (h16)v;
  }
}

// ---------------- WMMA GEMM: C[M,N] = A[M,K] * W[N,K]^T (+bias, act) ----------------
// block = 256 threads (8 waves) -> 128x64 macro-tile. A k-slab (128x32) staged in LDS
// via async global->LDS DMA; each wave computes a 16x64 subtile (4 WMMAs / k-step).
// ACT: 0=none, 1=relu, 2=silu
template<int ACT>
__global__ __launch_bounds__(256) void k_gemm(
    const h16* __restrict__ A, long bsA,
    const h16* __restrict__ W, long bsW,
    const float* __restrict__ bias,
    float* __restrict__ Cf, long bsCf,
    h16* __restrict__ Ch, long bsCh,
    int M, int N, int K) {
  __shared__ h16 As[128 * APITCH];
  int z = blockIdx.z;
  const h16* Ap = A + (long)z * bsA;
  const h16* Wp = W + (long)z * bsW;
  int tid = threadIdx.x;
  int wave = tid >> 5, lane = tid & 31;
  int hs = lane >> 4, l16 = lane & 15;
  int mblk = blockIdx.y << 7;   // 128-row macro tile
  int n0 = blockIdx.x << 6;
  int m0 = wave << 4;           // wave's rows within macro tile
  v8f acc[4] = {};
  for (int kk = 0; kk < K; kk += 32) {
    __syncthreads();            // protect previous iteration's LDS reads
    // stage A[mblk..mblk+127][kk..kk+31] -> LDS (512 x 16B chunks, 2 per thread)
    for (int c = tid; c < 512; c += 256) {
      int row = c >> 2, part = c & 3;
      lds_copy16(Ap + (long)(mblk + row) * K + kk + part * 8,
                 &As[row * APITCH + part * 8]);
    }
    lds_copy_join();
    __syncthreads();
    const h16* arow = &As[(m0 + l16) * APITCH];
    v8h a0 = *(const v8h*)(arow + 16 * hs);
    v8h a1 = *(const v8h*)(arow + 16 * hs + 8);
    v16h a;
#pragma unroll
    for (int i = 0; i < 8; ++i) { a[i] = a0[i]; a[i + 8] = a1[i]; }
#pragma unroll
    for (int j = 0; j < 4; ++j) {
      v16h bm = *(const v16h*)(Wp + (long)(n0 + j * 16 + l16) * K + kk + 16 * hs);
      acc[j] = __builtin_amdgcn_wmma_f32_16x16x32_f16(false, a, false, bm,
                                                      (short)0, acc[j], false, false);
    }
  }
#pragma unroll
  for (int j = 0; j < 4; ++j) {
    int col = n0 + j * 16 + l16;
    float bv = bias ? bias[col] : 0.f;
#pragma unroll
    for (int r = 0; r < 8; ++r) {
      int row = mblk + m0 + r + 8 * hs;
      float v = acc[j][r] + bv;
      if (ACT == 1) v = v > 0.f ? v : 0.f;
      if (ACT == 2) v = v / (1.f + __expf(-v));
      long idx = (long)row * N + col;
      if (Cf) Cf[(long)z * bsCf + idx] = v;
      if (Ch) Ch[(long)z * bsCh + idx] = (h16)v;
    }
  }
}

// ---------------- QKV repack: qkv[M,1536] -> q,k[BH,S,64] (q pre-scaled), vT[BH,64,S] ----
__global__ void k_qkv_repack(const h16* __restrict__ qkv, h16* __restrict__ q,
                             h16* __restrict__ k, h16* __restrict__ vT) {
  long i = (long)blockIdx.x * blockDim.x + threadIdx.x; // over MTOT*DD
  int d = (int)(i & 63);
  int h = (int)((i >> 6) & 7);
  long bs = i >> 9;
  int s = (int)(bs & (SMAX - 1));
  int b = (int)(bs >> 8);
  const h16* base = qkv + bs * (3 * DD);
  h16 qv = base[h * HDIM + d];
  h16 kv = base[DD + h * HDIM + d];
  h16 vv = base[2 * DD + h * HDIM + d];
  long bh = (long)(b * HH + h);
  q[(bh * SMAX + s) * HDIM + d] = (h16)((float)qv * 0.125f); // 1/sqrt(64)
  k[(bh * SMAX + s) * HDIM + d] = kv;
  vT[(bh * HDIM + d) * SMAX + s] = vv;
}

// ---------------- fused scores + masked softmax ----------------
// grid (SMAX/16, H, B), block 128 (4 waves). K=64, full 16x256 score block in LDS.
__global__ void k_scores_softmax(const h16* __restrict__ q, const h16* __restrict__ kx,
                                 const int* __restrict__ counts, h16* __restrict__ attn) {
  __shared__ float sc[16][SMAX + 1];
  int b = blockIdx.z, h = blockIdx.y;
  int m0 = blockIdx.x << 4;
  int tid = threadIdx.x;
  int wave = tid >> 5, lane = tid & 31;
  int hs = lane >> 4, l16 = lane & 15;
  long base = (long)(b * HH + h) * SMAX * HDIM;
  const h16* qrow = q + base + (long)(m0 + l16) * HDIM;
  int n0 = wave << 6;
  v8f acc[4] = {};
  for (int kk = 0; kk < HDIM; kk += 32) {
    v8h a0 = *(const v8h*)(qrow + kk + 16 * hs);
    v8h a1 = *(const v8h*)(qrow + kk + 16 * hs + 8);
    v16h a;
#pragma unroll
    for (int i = 0; i < 8; ++i) { a[i] = a0[i]; a[i + 8] = a1[i]; }
#pragma unroll
    for (int j = 0; j < 4; ++j) {
      v16h bm = *(const v16h*)(kx + base + (long)(n0 + j * 16 + l16) * HDIM + kk + 16 * hs);
      acc[j] = __builtin_amdgcn_wmma_f32_16x16x32_f16(false, a, false, bm,
                                                      (short)0, acc[j], false, false);
    }
  }
#pragma unroll
  for (int j = 0; j < 4; ++j)
#pragma unroll
    for (int r = 0; r < 8; ++r)
      sc[r + 8 * hs][n0 + j * 16 + l16] = acc[j][r];
  __syncthreads();
  int cnt = counts[b];
  if (cnt > SMAX) cnt = SMAX;
  if (tid < 16) {
    int r = tid, mrow = m0 + r;
    long orow = ((long)(b * HH + h) * SMAX + mrow) * SMAX;
    if (mrow >= cnt || cnt <= 0) {
      for (int c = 0; c < SMAX; ++c) attn[orow + c] = (h16)0.f;
    } else {
      float mx = -3.4e38f;
      for (int c = 0; c < cnt; ++c) { float v = sc[r][c]; mx = v > mx ? v : mx; }
      float sum = 0.f;
      for (int c = 0; c < cnt; ++c) { float e = __expf(sc[r][c] - mx); sc[r][c] = e; sum += e; }
      float inv = 1.f / sum;
      for (int c = 0; c < SMAX; ++c)
        attn[orow + c] = (h16)((c < cnt) ? sc[r][c] * inv : 0.f);
    }
  }
}

// o[BH,S,64] -> o512[(b*S+s)*512 + h*64 + d]
__global__ void k_o_repack(const h16* __restrict__ o, h16* __restrict__ o512) {
  long i = (long)blockIdx.x * blockDim.x + threadIdx.x; // over MTOT*DD
  int d = (int)(i & 63);
  int h = (int)((i >> 6) & 7);
  long bs = i >> 9;
  int s = (int)(bs & (SMAX - 1));
  int b = (int)(bs >> 8);
  o512[i] = o[((long)(b * HH + h) * SMAX + s) * HDIM + d];
}

// out = LN(res + add) * g + b  -> f32 and f16 copies. one block per row, 256 threads.
__global__ void k_residual_ln(const float* __restrict__ res, const float* __restrict__ add,
                              const float* __restrict__ g, const float* __restrict__ bb,
                              float* __restrict__ outF, h16* __restrict__ outH) {
  __shared__ float red[256];
  long row = blockIdx.x;
  int t = threadIdx.x;
  const float* r0 = res + row * DD;
  const float* a0 = add + row * DD;
  float v1 = r0[t] + a0[t];
  float v2 = r0[t + 256] + a0[t + 256];
  red[t] = v1 + v2;
  __syncthreads();
  for (int off = 128; off > 0; off >>= 1) { if (t < off) red[t] += red[t + off]; __syncthreads(); }
  float mu = red[0] / (float)DD;
  __syncthreads();
  float d1 = v1 - mu, d2 = v2 - mu;
  red[t] = d1 * d1 + d2 * d2;
  __syncthreads();
  for (int off = 128; off > 0; off >>= 1) { if (t < off) red[t] += red[t + off]; __syncthreads(); }
  float rstd = rsqrtf(red[0] / (float)DD + 1e-5f);
  float o1 = d1 * rstd * g[t] + bb[t];
  float o2 = d2 * rstd * g[t + 256] + bb[t + 256];
  outF[row * DD + t] = o1;
  outF[row * DD + t + 256] = o2;
  outH[row * DD + t] = (h16)o1;
  outH[row * DD + t + 256] = (h16)o2;
}

// masked mean pool per graph
__global__ void k_pool(const float* __restrict__ hpF, const int* __restrict__ counts,
                       h16* __restrict__ pooledH) {
  int b = blockIdx.x, t = threadIdx.x;
  int cnt = counts[b];
  int lim = cnt < SMAX ? cnt : SMAX;
  for (int d = t; d < DD; d += 256) {
    float s = 0.f;
    for (int ss = 0; ss < lim; ++ss) s += hpF[((long)b * SMAX + ss) * DD + d];
    pooledH[b * DD + d] = (h16)(cnt > 0 ? s / (float)cnt : 0.f);
  }
}

// y[b] = act[b,:] . w2 + b2   (one wave per graph)
__global__ void k_head2(const h16* __restrict__ act, const float* __restrict__ w2,
                        const float* __restrict__ b2, float* __restrict__ out) {
  int b = blockIdx.x, lane = threadIdx.x;
  float s = 0.f;
  for (int d = lane; d < DD; d += 32) s += (float)act[b * DD + d] * w2[d];
  for (int off = 16; off > 0; off >>= 1) s += __shfl_xor(s, off, 32);
  if (lane == 0) out[b] = s + b2[0];
}

// ---------------- launch ----------------
extern "C" void kernel_launch(void* const* d_in, const int* in_sizes, int n_in,
                              void* d_out, int out_size, void* d_ws, size_t ws_size,
                              hipStream_t stream) {
  (void)in_sizes; (void)n_in; (void)out_size; (void)ws_size;
  const float* node_features = (const float*)d_in[0];
  const int*   batch  = (const int*)d_in[3];
  const float* emb_w  = (const float*)d_in[4];
  const float* emb_b  = (const float*)d_in[5];
  const float* ipw    = (const float*)d_in[6];
  const float* ipb    = (const float*)d_in[7];
  const float* ow     = (const float*)d_in[8];
  const float* ob     = (const float*)d_in[9];
  const float* ln1g   = (const float*)d_in[10];
  const float* ln1b   = (const float*)d_in[11];
  const float* f1w    = (const float*)d_in[12];
  const float* f1b    = (const float*)d_in[13];
  const float* f2w    = (const float*)d_in[14];
  const float* f2b    = (const float*)d_in[15];
  const float* ln2g   = (const float*)d_in[16];
  const float* ln2b   = (const float*)d_in[17];
  const float* hw1    = (const float*)d_in[18];
  const float* hb1    = (const float*)d_in[19];
  const float* hw2    = (const float*)d_in[20];
  const float* hb2    = (const float*)d_in[21];
  float* y = (float*)d_out;

  // ---- workspace carve (bump allocator, 256B aligned) ----
  char* p = (char*)d_ws;
  auto alloc = [&](size_t bytes) -> char* {
    char* r = p; p += (bytes + 255) & ~(size_t)255; return r;
  };
  int* counts = (int*)alloc(BG * 4);
  int* starts = (int*)alloc(BG * 4);
  int* pos    = (int*)alloc((size_t)NN * 4);
  h16* embw16 = (h16*)alloc((size_t)DD * DINP * 2);
  h16* inw16  = (h16*)alloc((size_t)LL * 3 * DD * DD * 2);
  h16* outw16 = (h16*)alloc((size_t)LL * DD * DD * 2);
  h16* f1w16  = (h16*)alloc((size_t)LL * DFFN * DD * 2);
  h16* f2w16  = (h16*)alloc((size_t)LL * DD * DFFN * 2);
  h16* hw16   = (h16*)alloc((size_t)DD * DD * 2);
  h16* nf16   = (h16*)alloc((size_t)NN * DINP * 2);
  float* hpF  = (float*)alloc((size_t)MTOT * DD * 4);
  h16*   hpH  = (h16*)alloc((size_t)MTOT * DD * 2);
  // region A: reused (embed out f32 | qkv f16 | attn f16 | ffn-hidden f16)
  char* regA  = alloc((size_t)BG * HH * SMAX * SMAX * 2); // 128 MiB max
  // region B: q, k, vT
  h16* qB = (h16*)alloc((size_t)3 * BG * HH * SMAX * HDIM * 2);
  h16* kB = qB + (size_t)BG * HH * SMAX * HDIM;
  h16* vT = kB + (size_t)BG * HH * SMAX * HDIM;
  h16* oBH   = (h16*)alloc((size_t)BG * HH * SMAX * HDIM * 2);
  h16* o512  = (h16*)alloc((size_t)MTOT * DD * 2);
  float* eF  = (float*)alloc((size_t)MTOT * DD * 4); // proj / ffn-out f32
  float* xF  = (float*)alloc((size_t)MTOT * DD * 4);
  h16*   xH  = (h16*)alloc((size_t)MTOT * DD * 2);
  h16* pooledH = (h16*)alloc((size_t)BG * DD * 2);
  h16* actH    = (h16*)alloc((size_t)BG * DD * 2);

  float* hF   = (float*)regA; // embed output, N x 512
  h16*   qkvH = (h16*)regA;
  h16*   attnH= (h16*)regA;
  h16*   ffnH = (h16*)regA;

  // ---- meta ----
  k_fill_u32<<<1, 256, 0, stream>>>((uint32_t*)counts, 0u, BG);
  k_counts<<<cdiv(NN, 256), 256, 0, stream>>>(batch, counts);
  k_starts<<<1, 32, 0, stream>>>(counts, starts);
  k_pos<<<cdiv(NN, 256), 256, 0, stream>>>(batch, starts, pos);

  // ---- weight f16 conversion ----
  k_pad96<<<cdiv((long)DD * DINP, 256), 256, 0, stream>>>(emb_w, embw16, DD);
  k_pad96<<<cdiv((long)NN * DINP, 256), 256, 0, stream>>>(node_features, nf16, NN);
  k_cvt_f32_f16<<<cdiv((long)LL * 3 * DD * DD, 256), 256, 0, stream>>>(ipw, inw16, (long)LL * 3 * DD * DD);
  k_cvt_f32_f16<<<cdiv((long)LL * DD * DD, 256), 256, 0, stream>>>(ow, outw16, (long)LL * DD * DD);
  k_cvt_f32_f16<<<cdiv((long)LL * DFFN * DD, 256), 256, 0, stream>>>(f1w, f1w16, (long)LL * DFFN * DD);
  k_cvt_f32_f16<<<cdiv((long)LL * DD * DFFN, 256), 256, 0, stream>>>(f2w, f2w16, (long)LL * DD * DFFN);
  k_cvt_f32_f16<<<cdiv((long)DD * DD, 256), 256, 0, stream>>>(hw1, hw16, (long)DD * DD);

  // ---- zero padded activation (required: pad rows must stay finite) ----
  k_fill_u32<<<cdiv((long)MTOT * DD, 256), 256, 0, stream>>>((uint32_t*)hpF, 0u, (long)MTOT * DD);
  k_fill_u32<<<cdiv((long)MTOT * DD / 2, 256), 256, 0, stream>>>((uint32_t*)hpH, 0u, (long)MTOT * DD / 2);

  // ---- embedding: h = nf @ emb_w^T + emb_b  (M=NN, N=512, K=96) ----
  k_gemm<0><<<dim3(DD / 64, NN / 128, 1), 256, 0, stream>>>(
      nf16, 0, embw16, 0, emb_b, hF, 0, (h16*)nullptr, 0, NN, DD, DINP);
  k_scatter<<<NN, 256, 0, stream>>>(hF, batch, pos, hpF, hpH);

  long repack_total = (long)MTOT * DD;
  for (int l = 0; l < LL; ++l) {
    // QKV: (M, 1536) = hpH @ in_proj^T + b
    k_gemm<0><<<dim3(3 * DD / 64, MTOT / 128, 1), 256, 0, stream>>>(
        hpH, 0, inw16 + (size_t)l * 3 * DD * DD, 0, ipb + (size_t)l * 3 * DD,
        (float*)nullptr, 0, qkvH, 0, MTOT, 3 * DD, DD);
    k_qkv_repack<<<cdiv(repack_total, 256), 256, 0, stream>>>(qkvH, qB, kB, vT);
    // scores + softmax -> attn f16 (overwrites qkv region; qkv is dead)
    k_scores_softmax<<<dim3(SMAX / 16, HH, BG), 128, 0, stream>>>(qB, kB, counts, attnH);
    // o = attn @ v : batched (BH), A=attn(S,S), W=vT(64,S)
    k_gemm<0><<<dim3(1, SMAX / 128, BG * HH), 256, 0, stream>>>(
        attnH, (long)SMAX * SMAX, vT, (long)HDIM * SMAX, (const float*)nullptr,
        (float*)nullptr, 0, oBH, (long)SMAX * HDIM, SMAX, HDIM, SMAX);
    k_o_repack<<<cdiv(repack_total, 256), 256, 0, stream>>>(oBH, o512);
    // out proj -> f32
    k_gemm<0><<<dim3(DD / 64, MTOT / 128, 1), 256, 0, stream>>>(
        o512, 0, outw16 + (size_t)l * DD * DD, 0, ob + (size_t)l * DD,
        eF, 0, (h16*)nullptr, 0, MTOT, DD, DD);
    // LN1(hp + proj) -> x
    k_residual_ln<<<MTOT, 256, 0, stream>>>(hpF, eF, ln1g + (size_t)l * DD,
                                            ln1b + (size_t)l * DD, xF, xH);
    // FFN1: relu(x @ w1^T + b1) -> f16 hidden (attn region now dead)
    k_gemm<1><<<dim3(DFFN / 64, MTOT / 128, 1), 256, 0, stream>>>(
        xH, 0, f1w16 + (size_t)l * DFFN * DD, 0, f1b + (size_t)l * DFFN,
        (float*)nullptr, 0, ffnH, 0, MTOT, DFFN, DD);
    // FFN2 -> f32
    k_gemm<0><<<dim3(DD / 64, MTOT / 128, 1), 256, 0, stream>>>(
        ffnH, 0, f2w16 + (size_t)l * DD * DFFN, 0, f2b + (size_t)l * DD,
        eF, 0, (h16*)nullptr, 0, MTOT, DD, DFFN);
    // LN2(x + ffn) -> hp (next layer input)
    k_residual_ln<<<MTOT, 256, 0, stream>>>(xF, eF, ln2g + (size_t)l * DD,
                                            ln2b + (size_t)l * DD, hpF, hpH);
  }

  // ---- pool + head ----
  k_pool<<<BG, 256, 0, stream>>>(hpF, counts, pooledH);
  k_gemm<2><<<dim3(DD / 64, BG / 128, 1), 256, 0, stream>>>(
      pooledH, 0, hw16, 0, hb1, (float*)nullptr, 0, actH, 0, BG, DD, DD);
  k_head2<<<BG, 32, 0, stream>>>(actH, hw2, hb2, y);
}
